// ConvModel_79559974191304
// MI455X (gfx1250) — 8-bit matrix operand order repaired and checked by reading, compiled, not run on hardware
//
#include <hip/hip_runtime.h>
#include <stdint.h>


typedef __attribute__((ext_vector_type(8))) int v8i;

#define QA_MAX 127.0f
#define QA_MIN -128.0f
#define QW_MAX 3.0f
#define QW_MIN -4.0f

__device__ __forceinline__ float waveMaxF(float v) {
  #pragma unroll
  for (int m = 16; m >= 1; m >>= 1) v = fmaxf(v, __shfl_xor(v, m, 32));
  return v;
}

__device__ __forceinline__ float blockReduceMaxF(float v) {
  __shared__ float smax[32];
  const int lane = threadIdx.x & 31;
  const int wid  = threadIdx.x >> 5;
  v = waveMaxF(v);
  if (lane == 0) smax[wid] = v;
  __syncthreads();
  const int nw = (blockDim.x + 31) >> 5;
  v = (threadIdx.x < (unsigned)nw) ? smax[threadIdx.x] : 0.0f;
  if (wid == 0) v = waveMaxF(v);
  return v;
}

__device__ __forceinline__ void atomicMaxAbs(float* addr, float v) {
  atomicMax(reinterpret_cast<unsigned int*>(addr), __float_as_uint(v));
}

__device__ __forceinline__ float mkScale(float absmax, float qmax) {
  return fmaxf(absmax / qmax, 1e-8f);
}
__device__ __forceinline__ float qclampA(float x) {
  return fminf(fmaxf(rintf(x), QA_MIN), QA_MAX);
}
__device__ __forceinline__ float qclampW(float x) {
  return fminf(fmaxf(rintf(x), QW_MIN), QW_MAX);
}


__global__ void init_scal_kernel(float* scal) {
  if (threadIdx.x < 16) scal[threadIdx.x] = 0.0f;
}

__global__ void absmax_kernel(const float* __restrict__ x, int n,
                              float* __restrict__ target) {
  float lmax = 0.0f;
  for (int i = blockIdx.x * blockDim.x + threadIdx.x; i < n;
       i += gridDim.x * blockDim.x)
    lmax = fmaxf(lmax, fabsf(x[i]));
  lmax = blockReduceMaxF(lmax);
  if (threadIdx.x == 0) atomicMaxAbs(target, lmax);
}


__global__ void __launch_bounds__(256)
conv1_max_kernel(const float* __restrict__ image, const float* __restrict__ w1,
                 const float* __restrict__ b1, float* __restrict__ scal) {
  __shared__ float P[30 * 30];
  __shared__ float QW[128 * 4];
  const int t = threadIdx.x;
  const int b = blockIdx.x;
  const float s_img = mkScale(scal[0], QA_MAX);
  const float s_w1  = mkScale(scal[1], QW_MAX);

  for (int i = t; i < 900; i += blockDim.x) P[i] = 0.0f;
  __syncthreads();
  const float* img = image + b * 784;
  for (int i = t; i < 784; i += blockDim.x) {
    int y = i / 28, x = i - y * 28;
    P[(y + 1) * 30 + (x + 1)] = qclampA(img[i] / s_img);
  }
  for (int i = t; i < 512; i += blockDim.x) QW[i] = qclampW(w1[i] / s_w1);
  __syncthreads();

  const float sp = s_img * s_w1;
  float lmax = 0.0f;
  for (int idx = t; idx < 128 * 29 * 29; idx += blockDim.x) {
    int o = idx / 841;
    int rem = idx - o * 841;
    int y = rem / 29;
    int x = rem - y * 29;
    const float* p = &P[y * 30 + x];
    const float* w = &QW[o * 4];
    float acc = p[0] * w[0] + p[1] * w[1] + p[30] * w[2] + p[31] * w[3];
    float c1 = acc * sp + b1[o];
    lmax = fmaxf(lmax, fabsf(c1));
  }
  lmax = blockReduceMaxF(lmax);
  if (t == 0) atomicMaxAbs(&scal[4], lmax);
}


template <bool WRITE_Q>
__global__ void __launch_bounds__(256)
conv2_kernel(const float* __restrict__ image, const float* __restrict__ w1,
             const float* __restrict__ b1, const float* __restrict__ w2,
             const float* __restrict__ b2, float* __restrict__ scal,
             int8_t* __restrict__ q_o2) {
  __shared__ float P[30 * 30];
  __shared__ float QW1[128 * 4];
  __shared__ float QW2[128 * 4];
  __shared__ float O1[29 * 29];
  const int t = threadIdx.x;
  const int b = blockIdx.x;

  const float s_img = mkScale(scal[0], QA_MAX);
  const float s_w1  = mkScale(scal[1], QW_MAX);
  const float s_w2  = mkScale(scal[2], QW_MAX);
  const float am_c1 = scal[4];
  const float s_c1  = mkScale(am_c1, QA_MAX);
  const float s_o1  = fmaxf(tanhf(am_c1) / QA_MAX, 1e-8f);
  const float am_c2 = scal[5];
  const float s_c2  = mkScale(am_c2, QA_MAX);
  const float s_o2  = fmaxf(tanhf(am_c2) / QA_MAX, 1e-8f);

  for (int i = t; i < 900; i += blockDim.x) P[i] = 0.0f;
  __syncthreads();
  const float* img = image + b * 784;
  for (int i = t; i < 784; i += blockDim.x) {
    int y = i / 28, x = i - y * 28;
    P[(y + 1) * 30 + (x + 1)] = qclampA(img[i] / s_img);
  }
  for (int i = t; i < 512; i += blockDim.x) {
    QW1[i] = qclampW(w1[i] / s_w1);
    QW2[i] = qclampW(w2[i] / s_w2);
  }
  __syncthreads();

  const float sp1 = s_img * s_w1;
  const float sp2 = s_o1 * s_w2;
  float lmax = 0.0f;

  for (int c = 0; c < 128; ++c) {
    const float w00 = QW1[c * 4], w01 = QW1[c * 4 + 1];
    const float w10 = QW1[c * 4 + 2], w11 = QW1[c * 4 + 3];
    const float bias1 = b1[c];
    for (int i = t; i < 841; i += blockDim.x) {
      int y = i / 29, x = i - y * 29;
      const float* p = &P[y * 30 + x];
      float c1 = (p[0] * w00 + p[1] * w01 + p[30] * w10 + p[31] * w11) * sp1 + bias1;
      float qc1 = qclampA(c1 / s_c1) * s_c1;
      float o1  = tanhf(qc1);
      O1[i] = qclampA(o1 / s_o1);
    }
    __syncthreads();
    if (t < 196) {
      int yy = t / 14, xx = t - yy * 14;
      const float* q = &O1[(2 * yy) * 29 + 2 * xx];
      const float v00 = QW2[c * 4], v01 = QW2[c * 4 + 1];
      const float v10 = QW2[c * 4 + 2], v11 = QW2[c * 4 + 3];
      float c2 = (q[0] * v00 + q[1] * v01 + q[29] * v10 + q[30] * v11) * sp2 + b2[c];
      if (WRITE_Q) {
        float qc2 = qclampA(c2 / s_c2) * s_c2;
        float o2  = tanhf(qc2);
        int qo2 = (int)qclampA(o2 / s_o2);
        q_o2[(size_t)b * 25088 + (size_t)c * 196 + t] = (int8_t)qo2;
      } else {
        lmax = fmaxf(lmax, fabsf(c2));
      }
    }
    __syncthreads();
  }

  if (!WRITE_Q) {
    lmax = blockReduceMaxF(lmax);
    if (t == 0) atomicMaxAbs(&scal[5], lmax);
  }
}


__global__ void pack_wfc_kernel(const float* __restrict__ w_fc,
                                const float* __restrict__ scal,
                                int* __restrict__ packedB) {
  const int tid = blockIdx.x * blockDim.x + threadIdx.x;
  if (tid >= 392 * 256) return;
  const int kb   = tid >> 8;
  const int rem  = tid & 255;
  const int lane = rem >> 3;
  const int v    = rem & 7;
  const int n    = lane & 15;
  const int hi   = lane >> 4;
  const int kbase = kb * 64 + (v >> 1) * 16 + hi * 8 + (v & 1) * 4;
  const float s_wfc = mkScale(scal[3], QW_MAX);
  unsigned int dw = 0;
  #pragma unroll
  for (int by = 0; by < 4; ++by) {
    int k = kbase + by;
    int q = 0;
    if (n < 10 && k < 25088)
      q = (int)qclampW(w_fc[n * 25088 + k] / s_wfc);
    dw |= ((unsigned int)(q & 0xFF)) << (8 * by);
  }
  packedB[tid] = (int)dw;
}


__device__ __forceinline__ v8i load_a_frag(const int8_t* ap) {
  uint2 a0 = *reinterpret_cast<const uint2*>(ap);
  uint2 a1 = *reinterpret_cast<const uint2*>(ap + 16);
  uint2 a2 = *reinterpret_cast<const uint2*>(ap + 32);
  uint2 a3 = *reinterpret_cast<const uint2*>(ap + 48);
  v8i a;
  a[0] = (int)a0.x; a[1] = (int)a0.y; a[2] = (int)a1.x; a[3] = (int)a1.y;
  a[4] = (int)a2.x; a[5] = (int)a2.y; a[6] = (int)a3.x; a[7] = (int)a3.y;
  return a;
}

__device__ __forceinline__ v8i load_b_frag(const int* bp32) {
  const int4* bp = reinterpret_cast<const int4*>(bp32);
  int4 bl = bp[0];
  int4 bh = bp[1];
  v8i b;
  b[0] = bl.x; b[1] = bl.y; b[2] = bl.z; b[3] = bl.w;
  b[4] = bh.x; b[5] = bh.y; b[6] = bh.z; b[7] = bh.w;
  return b;
}

__global__ void __launch_bounds__(256)
fc_wmma_kernel(const int8_t* __restrict__ q_o2, const int* __restrict__ packedB,
               const float* __restrict__ b_fc, float* __restrict__ scal,
               float* __restrict__ logits) {
  const int lane = threadIdx.x & 31;
  const int wave = threadIdx.x >> 5;
  const int mt   = blockIdx.x * 8 + wave;
  const int m    = lane & 15;
  const int hi   = lane >> 4;

  const int8_t* aBase = q_o2 + (size_t)(mt * 16 + m) * 25088 + hi * 8;
  const int*    bBase = packedB + lane * 8;

  v8i acc0 = {0, 0, 0, 0, 0, 0, 0, 0};
  v8i acc1 = {0, 0, 0, 0, 0, 0, 0, 0};
  v8i acc2 = {0, 0, 0, 0, 0, 0, 0, 0};
  v8i acc3 = {0, 0, 0, 0, 0, 0, 0, 0};

  for (int kb = 0; kb < 392; kb += 4) {
    v8i a0 = load_a_frag(aBase + (kb + 0) * 64);
    v8i b0 = load_b_frag(bBase + (kb + 0) * 256);
    v8i a1 = load_a_frag(aBase + (kb + 1) * 64);
    v8i b1 = load_b_frag(bBase + (kb + 1) * 256);
    v8i a2 = load_a_frag(aBase + (kb + 2) * 64);
    v8i b2 = load_b_frag(bBase + (kb + 2) * 256);
    v8i a3 = load_a_frag(aBase + (kb + 3) * 64);
    v8i b3 = load_b_frag(bBase + (kb + 3) * 256);
    if (kb + 4 < 392) {
      __builtin_prefetch(aBase + (kb + 4) * 64, 0, 3);
      __builtin_prefetch(bBase + (kb + 4) * 256, 0, 3);
    }
    acc0 = __builtin_amdgcn_wmma_i32_16x16x64_iu8(true, a0, true, b0, acc0,
                                                  false, false);
    acc1 = __builtin_amdgcn_wmma_i32_16x16x64_iu8(true, a1, true, b1, acc1,
                                                  false, false);
    acc2 = __builtin_amdgcn_wmma_i32_16x16x64_iu8(true, a2, true, b2, acc2,
                                                  false, false);
    acc3 = __builtin_amdgcn_wmma_i32_16x16x64_iu8(true, a3, true, b3, acc3,
                                                  false, false);
  }

  const float s_o2  = fmaxf(tanhf(scal[5]) / QA_MAX, 1e-8f);
  const float s_wfc = mkScale(scal[3], QW_MAX);
  const float sc = s_o2 * s_wfc;
  const int n = m;
  float lmax = 0.0f;
  #pragma unroll
  for (int r = 0; r < 8; ++r) {
    if (n < 10) {
      int brow = mt * 16 + hi * 8 + r;
      int sum = acc0[r] + acc1[r] + acc2[r] + acc3[r];
      float lg = (float)sum * sc + b_fc[n];
      logits[brow * 10 + n] = lg;
      lmax = fmaxf(lmax, fabsf(lg));
    }
  }
  lmax = blockReduceMaxF(lmax);
  if (threadIdx.x == 0) atomicMaxAbs(&scal[6], lmax);
}


__global__ void final_fq_kernel(const float* __restrict__ logits,
                                const float* __restrict__ scal,
                                float* __restrict__ out) {
  const int i = blockIdx.x * blockDim.x + threadIdx.x;
  if (i >= 1024 * 10) return;
  const float s = mkScale(scal[6], QA_MAX);
  out[i] = qclampA(logits[i] / s) * s;
}


extern "C" void kernel_launch(void* const* d_in, const int* in_sizes, int n_in,
                              void* d_out, int out_size, void* d_ws,
                              size_t ws_size, hipStream_t stream) {
  const float* image = (const float*)d_in[0];
  const float* w1    = (const float*)d_in[1];
  const float* b1    = (const float*)d_in[2];
  const float* w2    = (const float*)d_in[3];
  const float* b2    = (const float*)d_in[4];
  const float* w_fc  = (const float*)d_in[5];
  const float* b_fc  = (const float*)d_in[6];
  float* out = (float*)d_out;

  char* ws = (char*)d_ws;
  float*  scal    = (float*)ws;
  int8_t* q_o2    = (int8_t*)(ws + 256);
  int*    packedB = (int*)(ws + 256 + 25690112);
  float*  logits  = (float*)(ws + 256 + 25690112 + 401408);

  init_scal_kernel<<<1, 32, 0, stream>>>(scal);

  absmax_kernel<<<256, 256, 0, stream>>>(image, 1024 * 784, &scal[0]);
  absmax_kernel<<<2,   256, 0, stream>>>(w1, 512, &scal[1]);
  absmax_kernel<<<2,   256, 0, stream>>>(w2, 512, &scal[2]);
  absmax_kernel<<<128, 256, 0, stream>>>(w_fc, 10 * 25088, &scal[3]);

  conv1_max_kernel<<<1024, 256, 0, stream>>>(image, w1, b1, scal);

  conv2_kernel<false><<<1024, 256, 0, stream>>>(image, w1, b1, w2, b2, scal,
                                                q_o2);
  conv2_kernel<true><<<1024, 256, 0, stream>>>(image, w1, b1, w2, b2, scal,
                                               q_o2);

  pack_wfc_kernel<<<392, 256, 0, stream>>>(w_fc, scal, packedB);

  fc_wmma_kernel<<<8, 256, 0, stream>>>(q_o2, packedB, b_fc, scal, logits);

  final_fq_kernel<<<40, 256, 0, stream>>>(logits, scal, out);
}
